// CrysformerLayer_69475390980549
// MI455X (gfx1250) — compile-verified, matrix operand order checked by reading, not run on hardware
//
#include <hip/hip_runtime.h>
#include <cstdint>
#include <cstddef>

// ---------------------------------------------------------------------------
// CDNA5 / gfx1250 types for WMMA bf16 path
// ---------------------------------------------------------------------------
typedef __attribute__((ext_vector_type(16))) __bf16 v16bf;
typedef __attribute__((ext_vector_type(8)))  __bf16 v8bf;
typedef __attribute__((ext_vector_type(8)))  float  v8f;

#define DIMC   128
#define HEADSC 8
#define HDIMC  16
#define SCALEC 0.25f      // HEAD_DIM^-0.5
#define EPSC   1e-5f
#define NTILE  8          // n-tiles per wave: 16x128 output strip, A reused 8x
#define BGRP   4          // B fragments loaded per clause (keeps VGPRs < spill)

// ---------------------------------------------------------------------------
// Generic bf16 WMMA GEMM: C[M,N] = act(A[M,K] @ B[K,N] + bias)
// A row-major bf16. B pre-packed into WMMA B-fragment order (see pack_b).
// One wave computes a 16 x (16*NTILE) strip. Per K-step, B fragments are
// consumed in groups of BGRP: one claused load group feeds BGRP back-to-back
// wmma, bounding live B registers to avoid scratch spills (64 acc + 32 B +
// 8 A VGPRs peak).  N must be a multiple of 16*NTILE (=128).
// act: 0=none 1=relu 2=tanh
// ---------------------------------------------------------------------------
__global__ void gemm_bf16_wmma(const __bf16* __restrict__ A,
                               const __bf16* __restrict__ Bp,
                               const float* __restrict__ bias,
                               float* __restrict__ Cf,
                               __bf16* __restrict__ Cb,
                               int M, int N, int K, int tiles, int act)
{
    int wave = blockIdx.x * (blockDim.x >> 5) + (threadIdx.x >> 5);
    if (wave >= tiles) return;                  // wave-uniform: EXEC stays full
    int lane    = threadIdx.x & 31;
    int tilesN  = N >> 4;
    int groupsN = tilesN / NTILE;
    int tm  = wave / groupsN;
    int tn0 = (wave - tm * groupsN) * NTILE;
    int hl = lane >> 4;                         // half-wave select (K-groups)
    int ml = lane & 15;                         // M index (A) / N index (B,C)

    int row = tm * 16 + ml;
    if (row >= M) row = M - 1;                  // clamp (no branch divergence)
    const __bf16* arow  = A + (size_t)row * K;
    const __bf16* bbase = Bp + ((size_t)tn0 * 32 + lane) * 16;

    v8f acc[NTILE];
#pragma unroll
    for (int j = 0; j < NTILE; ++j) acc[j] = (v8f){};

    int ksteps = K >> 5;
    for (int kt = 0; kt < ksteps; ++kt) {
        int kb = kt << 5;
        const __bf16* bp = bbase + (size_t)kt * tilesN * 512;

        // A 16x32 fragment: lanes 0-15 hold K {kb..kb+7, kb+16..kb+23},
        // lanes 16-31 hold K {kb+8..kb+15, kb+24..kb+31} (ISA 7.12.2)
        v8bf alo = *(const v8bf*)(arow + kb + hl * 8);
        v8bf ahi = *(const v8bf*)(arow + kb + 16 + hl * 8);
        if (kt + 1 < ksteps)
            __builtin_prefetch(arow + kb + 32, 0, 0);   // global_prefetch_b8
        v16bf a = __builtin_shufflevector(alo, ahi,
                   0,1,2,3,4,5,6,7,8,9,10,11,12,13,14,15);

#pragma unroll
        for (int g = 0; g < NTILE; g += BGRP) {
            // One claused group of 2*BGRP b128 loads ...
            v16bf bfrag[BGRP];
#pragma unroll
            for (int j = 0; j < BGRP; ++j) {
                v8bf blo = *(const v8bf*)(bp + (g + j) * 512);
                v8bf bhi = *(const v8bf*)(bp + (g + j) * 512 + 8);
                bfrag[j] = __builtin_shufflevector(blo, bhi,
                            0,1,2,3,4,5,6,7,8,9,10,11,12,13,14,15);
            }
            // ... amortized over BGRP back-to-back wmma
#pragma unroll
            for (int j = 0; j < BGRP; ++j) {
                acc[g + j] = __builtin_amdgcn_wmma_f32_16x16x32_bf16(
                                 false, a, false, bfrag[j], (short)0,
                                 acc[g + j], false, false);
            }
        }
    }

#pragma unroll
    for (int j = 0; j < NTILE; ++j) {
        int n  = (tn0 + j) * 16 + ml;
        float bv = bias ? bias[n] : 0.0f;
#pragma unroll
        for (int i = 0; i < 8; ++i) {
            int mm = tm * 16 + hl * 8 + i;      // C layout: lane half -> M+8
            if (mm >= M) continue;
            float v = acc[j][i] + bv;
            if (act == 1)      v = fmaxf(v, 0.0f);
            else if (act == 2) v = tanhf(v);
            size_t off = (size_t)mm * N + n;
            if (Cf) Cf[off] = v;
            if (Cb) Cb[off] = (__bf16)v;
        }
    }
}

// Pack a row-major f32 weight W[K,N] into the WMMA B-fragment stream:
// out[((kt*tilesN+tn)*32 + lane)*16 + elem]  (bf16)
__global__ void pack_b(const float* __restrict__ W, __bf16* __restrict__ out,
                       int K, int N)
{
    long long idx = (long long)blockIdx.x * blockDim.x + threadIdx.x;
    if (idx >= (long long)K * N) return;
    int elem = (int)(idx & 15);
    int lane = (int)((idx >> 4) & 31);
    long long tile = idx >> 9;
    int tilesN = N >> 4;
    int kt = (int)(tile / tilesN);
    int tn = (int)(tile - (long long)kt * tilesN);
    int hl = lane >> 4;
    int n  = tn * 16 + (lane & 15);
    int k  = kt * 32 + ((elem < 8) ? (hl * 8 + elem) : (16 + hl * 8 + (elem - 8)));
    out[idx] = (__bf16)W[(size_t)k * N + n];
}

__global__ void f2bf(const float* __restrict__ x, __bf16* __restrict__ y, long long n)
{
    long long i = (long long)blockIdx.x * blockDim.x + threadIdx.x;
    if (i < n) y[i] = (__bf16)x[i];
}

__global__ void fill_zero_u32(unsigned* __restrict__ p, long long n)
{
    long long i = (long long)blockIdx.x * blockDim.x + threadIdx.x;
    if (i < n) p[i] = 0u;
}

// ---------------------------------------------------------------------------
// Edge softmax helpers: order-preserving float<->uint encoding for atomicMax
// ---------------------------------------------------------------------------
__device__ __forceinline__ unsigned encF(float f) {
    unsigned u = __float_as_uint(f);
    return (u & 0x80000000u) ? ~u : (u | 0x80000000u);
}
__device__ __forceinline__ float decF(unsigned u) {
    return (u & 0x80000000u) ? __uint_as_float(u & 0x7fffffffu)
                             : __uint_as_float(~u);
}

// attention 1 logits: per (edge, head) dot(q[dst], k[e]) * SCALE, segment max
__global__ void attn1_logits(const float* __restrict__ q1,
                             const float* __restrict__ kv,
                             const int* __restrict__ dst,
                             float* __restrict__ logit,
                             unsigned* __restrict__ mEnc, int E)
{
    int idx = blockIdx.x * blockDim.x + threadIdx.x;
    if (idx >= E * HEADSC) return;
    int e = idx >> 3, hh = idx & 7;
    int dn = dst[e];
    const float* qr = q1 + (size_t)dn * DIMC + hh * HDIMC;
    const float* kr = kv + (size_t)e * (2 * DIMC) + hh * HDIMC;
    float s = 0.0f;
#pragma unroll
    for (int d = 0; d < HDIMC; ++d) s += qr[d] * kr[d];
    s *= SCALEC;
    logit[idx] = s;
    atomicMax(&mEnc[(size_t)dn * HEADSC + hh], encF(s));
}

// attention 2 logits: per (edge, head) sum of precomputed score row * SCALE
__global__ void attn2_logits(const float* __restrict__ score,
                             const int* __restrict__ dst,
                             float* __restrict__ logit,
                             unsigned* __restrict__ mEnc, int E)
{
    int idx = blockIdx.x * blockDim.x + threadIdx.x;
    if (idx >= E * HEADSC) return;
    int e = idx >> 3, hh = idx & 7;
    const float* sr = score + (size_t)e * DIMC + hh * HDIMC;
    float s = 0.0f;
#pragma unroll
    for (int d = 0; d < HDIMC; ++d) s += sr[d];
    s *= SCALEC;
    logit[idx] = s;
    atomicMax(&mEnc[(size_t)dst[e] * HEADSC + hh], encF(s));
}

// exp(logit - segmax) in place, segment-sum via atomics
__global__ void attn_exp(float* __restrict__ logit,
                         const unsigned* __restrict__ mEnc,
                         float* __restrict__ ssum,
                         const int* __restrict__ dst, int E)
{
    int idx = blockIdx.x * blockDim.x + threadIdx.x;
    if (idx >= E * HEADSC) return;
    int e = idx >> 3, hh = idx & 7;
    size_t so = (size_t)dst[e] * HEADSC + hh;
    float m = decF(mEnc[so]);
    if (!isfinite(m)) m = 0.0f;
    float ex = __expf(logit[idx] - m);
    logit[idx] = ex;
    atomicAdd(&ssum[so], ex);
}

// agg[dst] += (ex/ssum[dst]) * V[vrow, voff + d]; vrow = src ? src[e] : e
__global__ void attn_agg(const float* __restrict__ ex,
                         const float* __restrict__ ssum,
                         const float* __restrict__ V, int vstride, int voff,
                         const int* __restrict__ src,
                         const int* __restrict__ dst,
                         float* __restrict__ agg, int E)
{
    int idx = blockIdx.x * blockDim.x + threadIdx.x;
    if (idx >= E * DIMC) return;
    int e = idx >> 7, d = idx & 127, hh = d >> 4;
    int dn = dst[e];
    float w = ex[(size_t)e * HEADSC + hh] / ssum[(size_t)dn * HEADSC + hh];
    int vr = src ? src[e] : e;
    atomicAdd(&agg[(size_t)dn * DIMC + d],
              w * V[(size_t)vr * vstride + voff + d]);
}

// score[e,d] = k[src,d] * q[dst,d] * c[e,d] ; also emit bf16 copy for GEMM
__global__ void score_kernel(const float* __restrict__ qkv,
                             const float* __restrict__ c,
                             const int* __restrict__ src,
                             const int* __restrict__ dst,
                             float* __restrict__ sf,
                             __bf16* __restrict__ sb, int E)
{
    int idx = blockIdx.x * blockDim.x + threadIdx.x;
    if (idx >= E * DIMC) return;
    int e = idx >> 7, d = idx & 127;
    float k2 = qkv[(size_t)src[e] * (3 * DIMC) + DIMC + d];
    float q2 = qkv[(size_t)dst[e] * (3 * DIMC) + d];
    float s  = k2 * q2 * c[(size_t)e * DIMC + d];
    sf[idx] = s;
    sb[idx] = (__bf16)s;
}

// out[e,d] = mean over heads of score[e, h*16+d]
__global__ void score_mean(const float* __restrict__ sf,
                           float* __restrict__ out, int E)
{
    int idx = blockIdx.x * blockDim.x + threadIdx.x;
    if (idx >= E * HDIMC) return;
    int e = idx >> 4, d = idx & 15;
    float s = 0.0f;
#pragma unroll
    for (int hh = 0; hh < HEADSC; ++hh)
        s += sf[(size_t)e * DIMC + hh * HDIMC + d];
    out[idx] = s * (1.0f / HEADSC);
}

// out = LayerNorm(x + y) * g + beta ; one 128-thread block per row
__global__ void add_ln(const float* __restrict__ x, const float* __restrict__ y,
                       const float* __restrict__ g, const float* __restrict__ bt,
                       float* __restrict__ of, __bf16* __restrict__ ob)
{
    int r = blockIdx.x, i = threadIdx.x;
    size_t off = (size_t)r * DIMC + i;
    float v = x[off] + (y ? y[off] : 0.0f);
    __shared__ float sh[DIMC];
    sh[i] = v; __syncthreads();
    for (int s = DIMC / 2; s > 0; s >>= 1) {
        if (i < s) sh[i] += sh[i + s];
        __syncthreads();
    }
    float mu = sh[0] * (1.0f / DIMC);
    __syncthreads();
    float d = v - mu;
    sh[i] = d * d; __syncthreads();
    for (int s = DIMC / 2; s > 0; s >>= 1) {
        if (i < s) sh[i] += sh[i + s];
        __syncthreads();
    }
    float var = sh[0] * (1.0f / DIMC);
    float o = d * rsqrtf(var + EPSC) * g[i] + bt[i];
    of[off] = o;
    if (ob) ob[off] = (__bf16)o;
}

// ---------------------------------------------------------------------------
// Host driver
// ---------------------------------------------------------------------------
extern "C" void kernel_launch(void* const* d_in, const int* in_sizes, int n_in,
                              void* d_out, int out_size, void* d_ws, size_t ws_size,
                              hipStream_t stream)
{
    (void)n_in; (void)out_size; (void)ws_size;
    const float* h_in = (const float*)d_in[0];
    const float* e_in = (const float*)d_in[1];
    const int*   src  = (const int*)d_in[2];
    const int*   dst  = (const int*)d_in[3];
    const float* ca_q    = (const float*)d_in[4];
    const float* ca_kv   = (const float*)d_in[5];
    const float* ca_hp_w = (const float*)d_in[6];
    const float* ca_hp_b = (const float*)d_in[7];
    const float* co_qkv  = (const float*)d_in[8];
    const float* co_c    = (const float*)d_in[9];
    const float* co_hp_w = (const float*)d_in[10];
    const float* co_hp_b = (const float*)d_in[11];
    const float* co_ep_w = (const float*)d_in[12];
    const float* co_ep_b = (const float*)d_in[13];
    const float* m1_w1 = (const float*)d_in[14];
    const float* m1_b1 = (const float*)d_in[15];
    const float* m1_w2 = (const float*)d_in[16];
    const float* m1_b2 = (const float*)d_in[17];
    const float* m2_w1 = (const float*)d_in[18];
    const float* m2_b1 = (const float*)d_in[19];
    const float* m2_w2 = (const float*)d_in[20];
    const float* m2_b2 = (const float*)d_in[21];
    const float* ln_g[5] = { (const float*)d_in[22], (const float*)d_in[24],
                             (const float*)d_in[26], (const float*)d_in[28],
                             (const float*)d_in[30] };
    const float* ln_b[5] = { (const float*)d_in[23], (const float*)d_in[25],
                             (const float*)d_in[27], (const float*)d_in[29],
                             (const float*)d_in[31] };

    const int N = in_sizes[0] / DIMC;
    const int E = in_sizes[2];

    // --- workspace allocator (256B aligned bump) ---
    char* wsp = (char*)d_ws;
    auto alloc = [&](size_t bytes) -> void* {
        void* p = (void*)wsp;
        wsp += (bytes + 255) & ~(size_t)255;
        return p;
    };
    // packed bf16 weights
    __bf16* wq   = (__bf16*)alloc((size_t)128 * 128 * 2);
    __bf16* wkv  = (__bf16*)alloc((size_t)128 * 256 * 2);
    __bf16* whp  = (__bf16*)alloc((size_t)128 * 128 * 2);
    __bf16* wqkv = (__bf16*)alloc((size_t)128 * 384 * 2);
    __bf16* wc   = (__bf16*)alloc((size_t)128 * 128 * 2);
    __bf16* whp2 = (__bf16*)alloc((size_t)128 * 128 * 2);
    __bf16* wep  = (__bf16*)alloc((size_t)128 * 128 * 2);
    __bf16* w11  = (__bf16*)alloc((size_t)128 * 256 * 2);
    __bf16* w12  = (__bf16*)alloc((size_t)256 * 128 * 2);
    __bf16* w21  = (__bf16*)alloc((size_t)128 * 256 * 2);
    __bf16* w22  = (__bf16*)alloc((size_t)256 * 128 * 2);
    // activations / intermediates
    __bf16* h_bf  = (__bf16*)alloc((size_t)N * DIMC * 2);
    __bf16* e_bf  = (__bf16*)alloc((size_t)E * DIMC * 2);
    float*  q1    = (float*)alloc((size_t)N * DIMC * 4);
    float*  kv    = (float*)alloc((size_t)E * 2 * DIMC * 4);
    float*  logit = (float*)alloc((size_t)E * HEADSC * 4);
    unsigned* mEnc = (unsigned*)alloc((size_t)N * HEADSC * 4);
    float*  ssum  = (float*)alloc((size_t)N * HEADSC * 4);
    float*  hagg  = (float*)alloc((size_t)N * DIMC * 4);
    __bf16* hagg_bf = (__bf16*)alloc((size_t)N * DIMC * 2);
    float*  hagg2 = (float*)alloc((size_t)N * DIMC * 4);
    __bf16* hagg2_bf = (__bf16*)alloc((size_t)N * DIMC * 2);
    float*  nscr  = (float*)alloc((size_t)N * DIMC * 4);   // hp1 / hp2 / uh
    float*  h2_f  = (float*)alloc((size_t)N * DIMC * 4);
    __bf16* h2_b  = (__bf16*)alloc((size_t)N * DIMC * 2);
    float*  h3_f  = (float*)alloc((size_t)N * DIMC * 4);
    __bf16* h3_b  = (__bf16*)alloc((size_t)N * DIMC * 2);
    float*  qkv   = (float*)alloc((size_t)N * 3 * DIMC * 4);
    float*  cbuf  = (float*)alloc((size_t)E * DIMC * 4);
    float*  sc_f  = (float*)alloc((size_t)E * DIMC * 4);
    __bf16* sc_b  = (__bf16*)alloc((size_t)E * DIMC * 2);
    float*  e2_f  = (float*)alloc((size_t)E * DIMC * 4);
    __bf16* e2_b  = (__bf16*)alloc((size_t)E * DIMC * 2);
    float*  escr  = (float*)alloc((size_t)E * DIMC * 4);   // e_ / ue
    __bf16* th_b  = (__bf16*)alloc((size_t)N * 256 * 2);
    __bf16* te_b  = (__bf16*)alloc((size_t)E * 256 * 2);

    float* out_h  = (float*)d_out;
    float* out_e  = out_h + (size_t)N * DIMC;
    float* out_sm = out_e + (size_t)E * DIMC;

    auto blocks1d = [](long long n) { return (unsigned)((n + 255) / 256); };
    auto pack = [&](const float* W, __bf16* out, int K, int Nn) {
        long long n = (long long)K * Nn;
        pack_b<<<blocks1d(n), 256, 0, stream>>>(W, out, K, Nn);
    };
    auto gemm = [&](const __bf16* A, const __bf16* Bp, const float* bias,
                    float* Cf, __bf16* Cb, int M, int Nn, int K, int act) {
        int groupsN = (Nn >> 4) / NTILE;        // N multiple of 128 everywhere
        int tiles = ((M + 15) >> 4) * groupsN;
        gemm_bf16_wmma<<<(tiles + 7) / 8, 256, 0, stream>>>(
            A, Bp, bias, Cf, Cb, M, Nn, K, tiles, act);
    };
    auto zero = [&](void* p, long long words) {
        fill_zero_u32<<<blocks1d(words), 256, 0, stream>>>((unsigned*)p, words);
    };

    // ---- weight packing + input conversion ----
    pack(ca_q,   wq,   128, 128);
    pack(ca_kv,  wkv,  128, 256);
    pack(ca_hp_w,whp,  128, 128);
    pack(co_qkv, wqkv, 128, 384);
    pack(co_c,   wc,   128, 128);
    pack(co_hp_w,whp2, 128, 128);
    pack(co_ep_w,wep,  128, 128);
    pack(m1_w1,  w11,  128, 256);
    pack(m1_w2,  w12,  256, 128);
    pack(m2_w1,  w21,  128, 256);
    pack(m2_w2,  w22,  256, 128);
    f2bf<<<blocks1d((long long)N * DIMC), 256, 0, stream>>>(h_in, h_bf, (long long)N * DIMC);
    f2bf<<<blocks1d((long long)E * DIMC), 256, 0, stream>>>(e_in, e_bf, (long long)E * DIMC);

    // ---- cross-attention (edge k/v, node q) ----
    gemm(h_bf, wq,  nullptr, q1, nullptr, N, 128, 128, 0);       // q = h@ca_q
    gemm(e_bf, wkv, nullptr, kv, nullptr, E, 256, 128, 0);       // kv = e@ca_kv
    zero(mEnc, (long long)N * HEADSC);
    zero(ssum, (long long)N * HEADSC);
    zero(hagg, (long long)N * DIMC);
    attn1_logits<<<blocks1d((long long)E * HEADSC), 256, 0, stream>>>(q1, kv, dst, logit, mEnc, E);
    attn_exp<<<blocks1d((long long)E * HEADSC), 256, 0, stream>>>(logit, mEnc, ssum, dst, E);
    attn_agg<<<blocks1d((long long)E * DIMC), 256, 0, stream>>>(logit, ssum, kv, 2 * DIMC, DIMC, nullptr, dst, hagg, E);
    // hp1 = hagg @ ca_hp_w + ca_hp_b ; h = LN1(h + hp1)
    f2bf<<<blocks1d((long long)N * DIMC), 256, 0, stream>>>(hagg, hagg_bf, (long long)N * DIMC);
    gemm(hagg_bf, whp, ca_hp_b, nscr, nullptr, N, 128, 128, 0);
    add_ln<<<N, DIMC, 0, stream>>>(h_in, nscr, ln_g[0], ln_b[0], h2_f, h2_b);

    // ---- co-attention (node q/k/v, edge gate c) ----
    gemm(h2_b, wqkv, nullptr, qkv, nullptr, N, 384, 128, 0);     // qkv = h@co_qkv
    gemm(e_bf, wc,   nullptr, cbuf, nullptr, E, 128, 128, 2);    // c = tanh(e@co_c)
    score_kernel<<<blocks1d((long long)E * DIMC), 256, 0, stream>>>(qkv, cbuf, src, dst, sc_f, sc_b, E);
    zero(mEnc, (long long)N * HEADSC);
    zero(ssum, (long long)N * HEADSC);
    zero(hagg2, (long long)N * DIMC);
    attn2_logits<<<blocks1d((long long)E * HEADSC), 256, 0, stream>>>(sc_f, dst, logit, mEnc, E);
    attn_exp<<<blocks1d((long long)E * HEADSC), 256, 0, stream>>>(logit, mEnc, ssum, dst, E);
    attn_agg<<<blocks1d((long long)E * DIMC), 256, 0, stream>>>(logit, ssum, qkv, 3 * DIMC, 2 * DIMC, src, dst, hagg2, E);
    // h_ = hagg2 @ co_hp_w + b ; h = LN2(h2 + h_)
    f2bf<<<blocks1d((long long)N * DIMC), 256, 0, stream>>>(hagg2, hagg2_bf, (long long)N * DIMC);
    gemm(hagg2_bf, whp2, co_hp_b, nscr, nullptr, N, 128, 128, 0);
    add_ln<<<N, DIMC, 0, stream>>>(h2_f, nscr, ln_g[1], ln_b[1], h3_f, h3_b);
    // e_ = score @ co_ep_w + b ; e = LN3(e + e_)
    gemm(sc_b, wep, co_ep_b, escr, nullptr, E, 128, 128, 0);
    add_ln<<<E, DIMC, 0, stream>>>(e_in, escr, ln_g[2], ln_b[2], e2_f, e2_b);

    // ---- node MLP: h = LN4(h + relu(h@w1+b1)@w2 + b2) ----
    gemm(h3_b, w11, m1_b1, nullptr, th_b, N, 256, 128, 1);
    gemm(th_b, w12, m1_b2, nscr, nullptr, N, 128, 256, 0);
    add_ln<<<N, DIMC, 0, stream>>>(h3_f, nscr, ln_g[3], ln_b[3], out_h, nullptr);

    // ---- edge MLP: e = LN5(e + relu(e@w1+b1)@w2 + b2) ----
    gemm(e2_b, w21, m2_b1, nullptr, te_b, E, 256, 128, 1);
    gemm(te_b, w22, m2_b2, escr, nullptr, E, 128, 256, 0);
    add_ln<<<E, DIMC, 0, stream>>>(e2_f, escr, ln_g[4], ln_b[4], out_e, nullptr);

    // ---- score.mean(axis=1) ----
    score_mean<<<blocks1d((long long)E * HDIMC), 256, 0, stream>>>(sc_f, out_sm, E);
}
